// BasicBlock_23433341567004
// MI455X (gfx1250) — matrix operand order repaired and checked by reading, not compiled, not run on hardware
//
#include <hip/hip_runtime.h>

// ---------------- problem constants ----------------
#define Bn   4
#define Cn   128
#define Nn   32768
#define Kk   9          // KNN neighbors
#define KSk  5          // conv1d kernel size
#define KC2D ((Cn * Kk) / 32)    // 36 reduction chunks for gather-conv
#define KC1D ((Cn * KSk) / 32)   // 20 reduction chunks for weighted conv
#define EPSf 1e-5f

typedef __attribute__((ext_vector_type(16))) __bf16        v16bf;
typedef __attribute__((ext_vector_type(8)))  float         v8f;
typedef __attribute__((ext_vector_type(4)))  unsigned int  uint4v;

// ---------------- bf16 helpers (RNE) ----------------
__device__ __forceinline__ float bf2f(unsigned short h) {
    unsigned int u = ((unsigned int)h) << 16;
    float f; __builtin_memcpy(&f, &u, 4); return f;
}
__device__ __forceinline__ unsigned short f2bf(float f) {
    unsigned int u; __builtin_memcpy(&u, &f, 4);
    unsigned int r = u + 0x7FFFu + ((u >> 16) & 1u);
    return (unsigned short)(r >> 16);
}
// packed bf16 multiply (2 lanes per dword) — CDNA5 V_PK_MUL_BF16
__device__ __forceinline__ unsigned int pkmul_bf16(unsigned int a, unsigned int b) {
    unsigned int d;
    asm volatile("v_pk_mul_bf16 %0, %1, %2" : "=v"(d) : "v"(a), "v"(b));
    return d;
}

union Frag { v16bf v; unsigned short u[16]; unsigned int d[8]; uint4v q[2]; };

// ---------------- init: zero BN stats ----------------
__global__ void init_ws(float* __restrict__ stats) {
    int t = blockIdx.x * blockDim.x + threadIdx.x;
    if (t < 4 * 2 * Cn) stats[t] = 0.f;   // 4 stages x (sum,sumsq)
}

// ---------------- pack weights into WMMA A-fragment layout ----------------
// A 16x32 bf16 (this code's contiguous k order, not the ISA table order): lanes 0-15 rows M=0..15 (lanes 16-31 same rows),
// dword j, half h -> K = 2*j + h + (lane>=16 ? 16 : 0).
// Storage: [mt][kc][lane][8 dwords]. Reduction flattened K-MAJOR: r = k*Cn + c
// (so each 32-chunk has a single uniform k and contiguous c for the B side).
__global__ void pack_w(const float* __restrict__ w, unsigned int* __restrict__ apack,
                       int Kker, int KC, int total) {
    int dw = blockIdx.x * blockDim.x + threadIdx.x;
    if (dw >= total) return;
    int jd   = dw & 7;
    int lane = (dw >> 3) & 31;
    int kc   = (dw >> 8) % KC;
    int mt   = (dw >> 8) / KC;
    int o    = mt * 16 + (lane & 15);
    int kk0  = 2 * jd + ((lane >= 16) ? 16 : 0);
    int r0 = kc * 32 + kk0;
    int k0 = r0 >> 7, c0 = r0 & (Cn - 1);
    unsigned int lo = f2bf(w[((size_t)o * Cn + c0) * Kker + k0]);
    int r1 = r0 + 1;
    int k1 = r1 >> 7, c1 = r1 & (Cn - 1);
    unsigned int hi = f2bf(w[((size_t)o * Cn + c1) * Kker + k1]);
    apack[dw] = lo | (hi << 16);
}

// ---------------- x [B,C,N] f32  ->  xt [B,N,C] bf16 ----------------
__global__ void transpose_in(const float* __restrict__ x, unsigned short* __restrict__ xt) {
    __shared__ float t[32][33];
    int blk  = blockIdx.x;
    int tpb  = (Nn / 32) * (Cn / 32);
    int b    = blk / tpb;
    int rem  = blk % tpb;
    int n0   = (rem / (Cn / 32)) * 32;
    int c0   = (rem % (Cn / 32)) * 32;
    int tx = threadIdx.x & 31, ty = threadIdx.x >> 5;
    const float* xb = x + (size_t)b * Cn * Nn;
#pragma unroll
    for (int i = 0; i < 4; ++i) {
        int cl = ty + 8 * i;
        t[cl][tx] = xb[(size_t)(c0 + cl) * Nn + n0 + tx];   // coalesced over n
    }
    __syncthreads();
    unsigned short* xo = xt + (size_t)b * Nn * Cn;
#pragma unroll
    for (int i = 0; i < 4; ++i) {
        int nl = ty + 8 * i;
        xo[(size_t)(n0 + nl) * Cn + c0 + tx] = f2bf(t[tx][nl]);  // coalesced over c
    }
}

// ---------------- in-place elementwise: z = relu(z*scale + bias), bf16 ----------------
__global__ void apply_act(unsigned short* __restrict__ z, const float* __restrict__ sc,
                          const float* __restrict__ bs) {
    __shared__ float lsc[Cn], lbs[Cn];
    if (threadIdx.x < Cn) { lsc[threadIdx.x] = sc[threadIdx.x]; lbs[threadIdx.x] = bs[threadIdx.x]; }
    __syncthreads();
    size_t t = (size_t)blockIdx.x * blockDim.x + threadIdx.x;   // one uint4v = 8 channels
    int c0 = (int)((t & (Cn / 8 - 1)) * 8);
    uint4v* p = (uint4v*)z + t;
    uint4v v = *p;
    unsigned int d[4] = { v.x, v.y, v.z, v.w };
#pragma unroll
    for (int j = 0; j < 4; ++j) {
        int c = c0 + 2 * j;
        float lo = bf2f((unsigned short)(d[j] & 0xffffu));
        float hi = bf2f((unsigned short)(d[j] >> 16));
        lo = fmaxf(fmaf(lo, lsc[c],     lbs[c]),     0.f);
        hi = fmaxf(fmaf(hi, lsc[c + 1], lbs[c + 1]), 0.f);
        d[j] = (unsigned int)f2bf(lo) | ((unsigned int)f2bf(hi) << 16);
    }
    uint4v o = { d[0], d[1], d[2], d[3] };
    *p = o;
}

// ---------------- gather conv2d GEMM: z[b,n,o] = sum_{c,k} in[b,nbr(n,k),c]*W[o,c,k] ----
// src already activated (or raw x for stage 1). Emits BN stats of output.
__global__ __launch_bounds__(256)
void gemm_gather(const unsigned short* __restrict__ src, const int* __restrict__ eidx,
                 const uint4v* __restrict__ apack,
                 unsigned short* __restrict__ dst, float* __restrict__ stat) {
    __shared__ float lsum[Cn], lsq[Cn];
    int tid = threadIdx.x;
    if (tid < Cn) { lsum[tid] = 0.f; lsq[tid] = 0.f; }
    __syncthreads();

    int blk  = blockIdx.x;                    // B * N/128 blocks
    int b    = blk / (Nn / 128);
    int tile = blk % (Nn / 128);
    int lane = tid & 31;
    int nn   = tile * 128 + (tid >> 5) * 16 + (lane & 15);   // this lane's position
    int hi16 = (lane >= 16) ? 16 : 0;

    int nbr[Kk];
    const int* ip = eidx + ((size_t)b * Nn + nn) * Kk;
#pragma unroll
    for (int k = 0; k < Kk; ++k) nbr[k] = ip[k];

    const unsigned short* sb = src + (size_t)b * Nn * Cn;
    v8f zero = {0.f,0.f,0.f,0.f,0.f,0.f,0.f,0.f};
    v8f acc[8];
#pragma unroll
    for (int mt = 0; mt < 8; ++mt) acc[mt] = zero;

#pragma unroll
    for (int k = 0; k < Kk; ++k) {            // uniform k per 4 chunks (k-major flatten)
        const unsigned short* rowp = sb + (size_t)nbr[k] * Cn + hi16;
#pragma unroll
        for (int q = 0; q < 4; ++q) {         // c-base = q*32 + hi16
            Frag bf;
            const uint4v* bp = (const uint4v*)(rowp + q * 32);
            bf.q[0] = bp[0]; bf.q[1] = bp[1]; // 16 consecutive channels, fragment-ordered
            int kc = k * 4 + q;
            if (kc + 1 < KC2D)
                __builtin_prefetch(apack + (((size_t)kc + 1) * 32 + lane) * 2, 0, 0);
#pragma unroll
            for (int mt = 0; mt < 8; ++mt) {
                Frag af;
                const uint4v* ap = apack + (((size_t)mt * KC2D + kc) * 32 + lane) * 2;
                af.q[0] = ap[0]; af.q[1] = ap[1];
                acc[mt] = __builtin_amdgcn_wmma_f32_16x16x32_bf16(
                    false, af.v, false, bf.v, (short)0, acc[mt], false, false);
            }
        }
    }

    // D layout: lane = N col, vgpr v -> channel = mt*16 + v + (lane>=16 ? 8 : 0)
    int choff = (lane >= 16) ? 8 : 0;
    unsigned short* db = dst + ((size_t)b * Nn + nn) * Cn;
#pragma unroll
    for (int mt = 0; mt < 8; ++mt) {
        unsigned int dwv[4];
#pragma unroll
        for (int v = 0; v < 8; ++v) {
            float f = acc[mt][v];
            atomicAdd(&lsum[mt * 16 + choff + v], f);
            atomicAdd(&lsq [mt * 16 + choff + v], f * f);
            unsigned int bits = f2bf(f);
            if (v & 1) dwv[v >> 1] |= bits << 16; else dwv[v >> 1] = bits;
        }
        uint4v o4 = { dwv[0], dwv[1], dwv[2], dwv[3] };
        *(uint4v*)(db + mt * 16 + choff) = o4;   // 8 consecutive bf16 channels
    }
    __syncthreads();
    if (tid < Cn) { atomicAdd(&stat[tid], lsum[tid]); atomicAdd(&stat[Cn + tid], lsq[tid]); }
}

// ---------------- gaussian-weighted conv1d GEMM ----------------
// z[b,n,o] = sum_{c,k} u[b,n+k-2,c] * gw[b,k,n] * Wc[o,c,k]   (u pre-activated)
__global__ __launch_bounds__(256)
void wconv(const unsigned short* __restrict__ src, const float* __restrict__ coords,
           const uint4v* __restrict__ apack,
           unsigned short* __restrict__ dst, float* __restrict__ stat) {
    __shared__ float lsum[Cn], lsq[Cn];
    int tid = threadIdx.x;
    if (tid < Cn) { lsum[tid] = 0.f; lsq[tid] = 0.f; }
    __syncthreads();

    int blk  = blockIdx.x;
    int b    = blk / (Nn / 128);
    int tile = blk % (Nn / 128);
    int lane = tid & 31;
    int nn   = tile * 128 + (tid >> 5) * 16 + (lane & 15);
    int hi16 = (lane >= 16) ? 16 : 0;

    // gaussian window weights for this lane's position (sigma = 1)
    const float* cb = coords + (size_t)b * 3 * Nn;
    float x0 = cb[nn], x1 = cb[Nn + nn], x2 = cb[2 * Nn + nn];
    float gw[KSk];
#pragma unroll
    for (int k = 0; k < KSk; ++k) {
        int p = nn + k - 2;
        if (p < 0 || p >= Nn) { gw[k] = 0.f; }   // zero-padded x nullifies term
        else {
            float d0 = cb[p] - x0, d1 = cb[Nn + p] - x1, d2 = cb[2 * Nn + p] - x2;
            gw[k] = __expf(-0.5f * (d0 * d0 + d1 * d1 + d2 * d2));
        }
    }

    const unsigned short* sb = src + (size_t)b * Nn * Cn;
    v8f zero = {0.f,0.f,0.f,0.f,0.f,0.f,0.f,0.f};
    v8f acc[8];
#pragma unroll
    for (int mt = 0; mt < 8; ++mt) acc[mt] = zero;

#pragma unroll
    for (int k = 0; k < KSk; ++k) {           // uniform k per 4 chunks
        int p  = nn + k - 2;
        int pc = p < 0 ? 0 : (p >= Nn ? Nn - 1 : p);   // clamp; gw[k]==0 kills OOB
        const unsigned short* rowp = sb + (size_t)pc * Cn + hi16;
        unsigned int gsp = (unsigned int)f2bf(gw[k]);
        gsp |= gsp << 16;                     // packed bf16 splat of gw[k]
#pragma unroll
        for (int q = 0; q < 4; ++q) {
            Frag bf;
            const uint4v* bp = (const uint4v*)(rowp + q * 32);
            bf.q[0] = bp[0]; bf.q[1] = bp[1];
#pragma unroll
            for (int j = 0; j < 8; ++j) bf.d[j] = pkmul_bf16(bf.d[j], gsp);
            int kc = k * 4 + q;
            if (kc + 1 < KC1D)
                __builtin_prefetch(apack + (((size_t)kc + 1) * 32 + lane) * 2, 0, 0);
#pragma unroll
            for (int mt = 0; mt < 8; ++mt) {
                Frag af;
                const uint4v* ap = apack + (((size_t)mt * KC1D + kc) * 32 + lane) * 2;
                af.q[0] = ap[0]; af.q[1] = ap[1];
                acc[mt] = __builtin_amdgcn_wmma_f32_16x16x32_bf16(
                    false, af.v, false, bf.v, (short)0, acc[mt], false, false);
            }
        }
    }

    int choff = (lane >= 16) ? 8 : 0;
    unsigned short* db = dst + ((size_t)b * Nn + nn) * Cn;
#pragma unroll
    for (int mt = 0; mt < 8; ++mt) {
        unsigned int dwv[4];
#pragma unroll
        for (int v = 0; v < 8; ++v) {
            float f = acc[mt][v];
            atomicAdd(&lsum[mt * 16 + choff + v], f);
            atomicAdd(&lsq [mt * 16 + choff + v], f * f);
            unsigned int bits = f2bf(f);
            if (v & 1) dwv[v >> 1] |= bits << 16; else dwv[v >> 1] = bits;
        }
        uint4v o4 = { dwv[0], dwv[1], dwv[2], dwv[3] };
        *(uint4v*)(db + mt * 16 + choff) = o4;
    }
    __syncthreads();
    if (tid < Cn) { atomicAdd(&stat[tid], lsum[tid]); atomicAdd(&stat[Cn + tid], lsq[tid]); }
}

// ---------------- fold BN stats into per-channel scale/bias ----------------
__global__ void bnparams(const float* __restrict__ stat, const float* __restrict__ gamma,
                         const float* __restrict__ beta,
                         float* __restrict__ scale, float* __restrict__ bias) {
    int c = threadIdx.x;
    float cnt = (float)(Bn * Nn);
    float m = stat[c] / cnt;
    float v = stat[Cn + c] / cnt - m * m;
    float s = gamma[c] * rsqrtf(v + EPSf);
    scale[c] = s;
    bias[c]  = beta[c] - m * s;
}

// ---------------- final: out[b,c,n] = relu(bn(z)[b,n,c] + x[b,c,n]) ----------------
__global__ void finalize(const unsigned short* __restrict__ z, const float* __restrict__ x,
                         const float* __restrict__ sc, const float* __restrict__ bs,
                         float* __restrict__ out) {
    __shared__ float t[32][33];
    int blk  = blockIdx.x;
    int tpb  = (Nn / 32) * (Cn / 32);
    int b    = blk / tpb;
    int rem  = blk % tpb;
    int n0   = (rem / (Cn / 32)) * 32;
    int c0   = (rem % (Cn / 32)) * 32;
    int tx = threadIdx.x & 31, ty = threadIdx.x >> 5;
    const unsigned short* zb = z + (size_t)b * Nn * Cn;
#pragma unroll
    for (int i = 0; i < 4; ++i) {
        int nl = ty + 8 * i;
        int c  = c0 + tx;
        t[nl][tx] = fmaf(bf2f(zb[(size_t)(n0 + nl) * Cn + c]), sc[c], bs[c]);  // coalesced over c
    }
    __syncthreads();
    const float* xb = x + (size_t)b * Cn * Nn;
    float* ob = out + (size_t)b * Cn * Nn;
#pragma unroll
    for (int i = 0; i < 4; ++i) {
        int cl = ty + 8 * i;
        size_t off = (size_t)(c0 + cl) * Nn + n0 + tx;
        ob[off] = fmaxf(t[tx][cl] + xb[off], 0.f);   // coalesced over n
    }
}

// ---------------- host launcher ----------------
extern "C" void kernel_launch(void* const* d_in, const int* in_sizes, int n_in,
                              void* d_out, int out_size, void* d_ws, size_t ws_size,
                              hipStream_t stream) {
    const float* x      = (const float*)d_in[0];
    const int*   eidx   = (const int*)  d_in[1];
    const float* coords = (const float*)d_in[2];
    const float* w2d1 = (const float*)d_in[3];
    const float* g2d1 = (const float*)d_in[4];
    const float* b2d1 = (const float*)d_in[5];
    const float* wc1  = (const float*)d_in[6];
    const float* g1d1 = (const float*)d_in[7];
    const float* b1d1 = (const float*)d_in[8];
    const float* w2d2 = (const float*)d_in[9];
    const float* g2d2 = (const float*)d_in[10];
    const float* b2d2 = (const float*)d_in[11];
    const float* wc2  = (const float*)d_in[12];
    const float* g1d2 = (const float*)d_in[13];
    const float* b1d2 = (const float*)d_in[14];
    float* out = (float*)d_out;

    // workspace carve-up
    char* ws = (char*)d_ws;
    size_t off = 0;
    const size_t zbytes = (size_t)Bn * Nn * Cn * sizeof(unsigned short);  // 32 MB
    unsigned short* xt = (unsigned short*)(ws + off); off += zbytes;
    unsigned short* zA = (unsigned short*)(ws + off); off += zbytes;
    unsigned short* zB = (unsigned short*)(ws + off); off += zbytes;
    const int dw2d = 8 * KC2D * 32 * 8;   // dwords per packed gather-conv weight
    const int dw1d = 8 * KC1D * 32 * 8;   // dwords per packed weighted-conv weight
    unsigned int* ap1 = (unsigned int*)(ws + off); off += (size_t)dw2d * 4;
    unsigned int* ap2 = (unsigned int*)(ws + off); off += (size_t)dw2d * 4;
    unsigned int* wp1 = (unsigned int*)(ws + off); off += (size_t)dw1d * 4;
    unsigned int* wp2 = (unsigned int*)(ws + off); off += (size_t)dw1d * 4;
    float* stats  = (float*)(ws + off); off += 4 * 2 * Cn * sizeof(float);
    float* params = (float*)(ws + off); off += 8 * Cn * sizeof(float);
    float* sc1 = params;            float* bs1 = params + Cn;
    float* sc2 = params + 2 * Cn;   float* bs2 = params + 3 * Cn;
    float* sc3 = params + 4 * Cn;   float* bs3 = params + 5 * Cn;
    float* sc4 = params + 6 * Cn;   float* bs4 = params + 7 * Cn;

    init_ws<<<4, 256, 0, stream>>>(stats);
    pack_w<<<(dw2d + 255) / 256, 256, 0, stream>>>(w2d1, ap1, Kk,  KC2D, dw2d);
    pack_w<<<(dw2d + 255) / 256, 256, 0, stream>>>(w2d2, ap2, Kk,  KC2D, dw2d);
    pack_w<<<(dw1d + 255) / 256, 256, 0, stream>>>(wc1,  wp1, KSk, KC1D, dw1d);
    pack_w<<<(dw1d + 255) / 256, 256, 0, stream>>>(wc2,  wp2, KSk, KC1D, dw1d);

    int tgrid = Bn * (Nn / 32) * (Cn / 32);
    transpose_in<<<tgrid, 256, 0, stream>>>(x, xt);

    int ggrid = Bn * (Nn / 128);
    int agrid = (int)(((size_t)Bn * Nn * Cn / 8) / 256);   // apply_act grid

    // ---- sep_conv 1 ----
    gemm_gather<<<ggrid, 256, 0, stream>>>(xt, eidx, (const uint4v*)ap1, zA, stats);
    bnparams<<<1, Cn, 0, stream>>>(stats,           g2d1, b2d1, sc1, bs1);
    apply_act<<<agrid, 256, 0, stream>>>(zA, sc1, bs1);            // relu(bn(z1))
    wconv<<<ggrid, 256, 0, stream>>>(zA, coords, (const uint4v*)wp1, zB, stats + 2 * Cn);
    bnparams<<<1, Cn, 0, stream>>>(stats + 2 * Cn,  g1d1, b1d1, sc2, bs2);
    apply_act<<<agrid, 256, 0, stream>>>(zB, sc2, bs2);            // relu(bn(z2)) (act=True)
    // ---- sep_conv 2 ----
    gemm_gather<<<ggrid, 256, 0, stream>>>(zB, eidx, (const uint4v*)ap2, zA, stats + 4 * Cn);
    bnparams<<<1, Cn, 0, stream>>>(stats + 4 * Cn,  g2d2, b2d2, sc3, bs3);
    apply_act<<<agrid, 256, 0, stream>>>(zA, sc3, bs3);            // relu(bn(z3))
    wconv<<<ggrid, 256, 0, stream>>>(zA, coords, (const uint4v*)wp2, zB, stats + 6 * Cn);
    bnparams<<<1, Cn, 0, stream>>>(stats + 6 * Cn,  g1d2, b1d2, sc4, bs4);
    // ---- residual + relu, back to [B,C,N]; final BN folded in ----
    finalize<<<tgrid, 256, 0, stream>>>(zB, x, sc4, bs4, out);
}
